// SpatialAttentionBlock_43147241456012
// MI455X (gfx1250) — compile-verified, matrix operand order checked by reading, not run on hardware
//
#include <hip/hip_runtime.h>
#include <stdint.h>

typedef __attribute__((ext_vector_type(16))) __bf16 v16bf;
typedef __attribute__((ext_vector_type(8)))  float  v8f;

union FragBF {
  v16bf v;
  uint4 q[2];
  uint16_t h[16];
};

__device__ __forceinline__ uint16_t f2bf(float f) {
  union { float f; uint32_t u; } cv; cv.f = f;
  uint32_t r = cv.u + 0x7FFFu + ((cv.u >> 16) & 1u);
  return (uint16_t)(r >> 16);
}

// ---------------------------------------------------------------------------
// Tensor Data Mover: 2D tile load (256 rows x 32 bf16, row stride 4096 elems)
// Descriptor layout per CDNA5 ISA (D# groups 0/1).
// This toolchain exposes the 6-arg (clang-23 / therock-10.0) builtin form.
// ---------------------------------------------------------------------------
#if __has_builtin(__builtin_amdgcn_tensor_load_to_lds)
#define HAVE_TDM 1
typedef uint32_t u32x4 __attribute__((ext_vector_type(4)));
typedef int      i32x8 __attribute__((ext_vector_type(8)));
typedef int      i32x4 __attribute__((ext_vector_type(4)));

__device__ __forceinline__ void tdm_load_v_chunk(const uint16_t* gptr, uint32_t lds_off) {
  uint64_t ga = (uint64_t)(uintptr_t)gptr;
  u32x4 g0;
  g0[0] = 1u;                                   // count=1, user descriptor
  g0[1] = lds_off;                              // lds_addr (bytes)
  g0[2] = (uint32_t)ga;                         // global_addr[31:0]
  g0[3] = (uint32_t)(ga >> 32) | (2u << 30);    // global_addr[56:32] | type=2
  i32x8 g1;
  g1[0] = (int)(1u << 16);                      // data_size=1 -> 2 bytes/elem
  g1[1] = (int)(4096u << 16);                   // tensor_dim0=4096 (bits 79:48)
  g1[2] = (int)(256u << 16);                    // tensor_dim1=256  (bits 111:80)
  g1[3] = (int)(32u << 16);                     // tile_dim0=32     (bits 127:112)
  g1[4] = 256;                                  // tile_dim1=256, tile_dim2=0
  g1[5] = 4096;                                 // tensor_dim0_stride=4096 (48b)
  g1[6] = 0;                                    // stride hi | tensor_dim1_stride lo
  g1[7] = 0;
  __builtin_amdgcn_tensor_load_to_lds(g0, g1, (i32x4)0, (i32x4)0, (i32x8)0, 0);
}
#else
#define HAVE_TDM 0
#endif

// ---------------------------------------------------------------------------
// Kernel 1: q (1x3 conv, pad w) and k (3x1 conv, pad h), 32 out channels.
// Qt[b][n][32] bf16, Kt[b][n][32] bf16 (K-dim = C8 = 32 = one bf16 WMMA K).
// ---------------------------------------------------------------------------
#define CB 16
__global__ __launch_bounds__(256) void qk_conv_kernel(
    const float* __restrict__ x, const float* __restrict__ wq, const float* __restrict__ bq,
    const float* __restrict__ wk, const float* __restrict__ bk,
    uint16_t* __restrict__ Qt, uint16_t* __restrict__ Kt)
{
  const int h = blockIdx.x;       // 0..63
  const int b = blockIdx.y;       // 0..7
  const int t = threadIdx.x;      // 0..255
  __shared__ float xt[CB][3][64]; // [c][dy][w], rows h-1,h,h+1 (zero padded)

  const int o     = t & 31;       // out channel 0..31
  const int wbase = (t >> 5) * 8; // 8 w positions per thread

  float accq[8], acck[8];
  #pragma unroll
  for (int u = 0; u < 8; ++u) { accq[u] = 0.f; acck[u] = 0.f; }

  #pragma unroll 1
  for (int cc = 0; cc < 256; cc += CB) {
    __syncthreads();
    #pragma unroll
    for (int i = 0; i < (CB * 3 * 64) / 256; ++i) {
      int e = t + 256 * i;
      int c = e / 192, rem = e % 192;
      int dy = rem / 64, w = rem % 64;
      int hh = h + dy - 1;
      float v = 0.f;
      if (hh >= 0 && hh < 64)
        v = x[(((size_t)b * 256 + cc + c) * 64 + hh) * 64 + w];
      xt[c][dy][w] = v;
    }
    __syncthreads();
    #pragma unroll 4
    for (int c = 0; c < CB; ++c) {
      #pragma unroll
      for (int tap = 0; tap < 3; ++tap) {
        float wqv = wq[((o * 256) + cc + c) * 3 + tap]; // (32,256,1,3)
        float wkv = wk[((o * 256) + cc + c) * 3 + tap]; // (32,256,3,1)
        #pragma unroll
        for (int u = 0; u < 8; ++u) {
          int w  = wbase + u;
          int ww = w + tap - 1;                         // horizontal tap (q)
          float xq = (ww >= 0 && ww < 64) ? xt[c][1][ww] : 0.f;
          accq[u] = fmaf(wqv, xq, accq[u]);
          acck[u] = fmaf(wkv, xt[c][tap][w], acck[u]);  // vertical tap (k)
        }
      }
    }
  }
  float qb = bq[o], kb = bk[o];
  #pragma unroll
  for (int u = 0; u < 8; ++u) {
    int n = h * 64 + wbase + u;
    size_t idx = ((size_t)b * 4096 + n) * 32 + o;
    Qt[idx] = f2bf(accq[u] + qb);
    Kt[idx] = f2bf(acck[u] + kb);
  }
}

// ---------------------------------------------------------------------------
// Kernel 2: v = 1x1 conv as WMMA GEMM: Vt[b][c][n] = bv[c] + wv[c][:].x[b][:][n]
// ---------------------------------------------------------------------------
__global__ __launch_bounds__(256) void v_conv_wmma_kernel(
    const float* __restrict__ x, const float* __restrict__ wv, const float* __restrict__ bv,
    uint16_t* __restrict__ Vt)
{
  const int n0   = blockIdx.x * 32;
  const int b    = blockIdx.z;
  const int t    = threadIdx.x;
  const int wave = t >> 5;
  const int lane = t & 31;
  const int half = lane >> 4;
  const int l16  = lane & 15;
  const int c0   = blockIdx.y * 128 + wave * 16;

  __shared__ __align__(16) uint16_t xT[32][32]; // [n_local][c'_local] bf16

  v8f acc0 = {}, acc1 = {};

  #pragma unroll 1
  for (int cc = 0; cc < 256; cc += 32) {
    __syncthreads();
    { // stage x[b][cc..cc+32][n0..n0+32] transposed
      int cl = t >> 3;
      int nl = (t & 7) * 4;
      const float4 xv = *(const float4*)&x[((size_t)b * 256 + cc + cl) * 4096 + n0 + nl];
      xT[nl + 0][cl] = f2bf(xv.x);
      xT[nl + 1][cl] = f2bf(xv.y);
      xT[nl + 2][cl] = f2bf(xv.z);
      xT[nl + 3][cl] = f2bf(xv.w);
    }
    __syncthreads();

    FragBF a; // wv rows, on-the-fly fp32->bf16
    {
      const float* wr = wv + (size_t)(c0 + l16) * 256 + cc;
      #pragma unroll
      for (int e = 0; e < 8; ++e) {
        a.h[e]     = f2bf(wr[half * 8 + e]);
        a.h[e + 8] = f2bf(wr[16 + half * 8 + e]);
      }
    }
    FragBF b0, b1;
    b0.q[0] = *(const uint4*)&xT[l16][half * 8];
    b0.q[1] = *(const uint4*)&xT[l16][16 + half * 8];
    b1.q[0] = *(const uint4*)&xT[16 + l16][half * 8];
    b1.q[1] = *(const uint4*)&xT[16 + l16][16 + half * 8];

    acc0 = __builtin_amdgcn_wmma_f32_16x16x32_bf16(false, a.v, false, b0.v, (short)0, acc0, false, false);
    acc1 = __builtin_amdgcn_wmma_f32_16x16x32_bf16(false, a.v, false, b1.v, (short)0, acc1, false, false);
  }
  #pragma unroll
  for (int r = 0; r < 8; ++r) {
    int c = c0 + half * 8 + r;
    float bias = bv[c];
    Vt[((size_t)b * 256 + c) * 4096 + n0 + l16]      = f2bf(acc0[r] + bias);
    Vt[((size_t)b * 256 + c) * 4096 + n0 + 16 + l16] = f2bf(acc1[r] + bias);
  }
}

// ---------------------------------------------------------------------------
// Kernel 3: fused flash attention + residual, TDM double-buffered V staging.
// ---------------------------------------------------------------------------
#define NCHUNK 128  // 4096 / 32

__global__ __launch_bounds__(256) void attn_kernel(
    const float* __restrict__ x, const float* __restrict__ gamma_p,
    const uint16_t* __restrict__ Qt, const uint16_t* __restrict__ Kt,
    const uint16_t* __restrict__ Vt, float* __restrict__ out)
{
  const int b    = blockIdx.y;
  const int i0   = blockIdx.x * 128;
  const int t    = threadIdx.x;
  const int wave = t >> 5;
  const int lane = t & 31;
  const int half = lane >> 4;
  const int l16  = lane & 15;

  __shared__ __align__(16) uint16_t Vs[2][256][32]; // double-buffered V, 32 KB
  __shared__ __align__(16) uint16_t Ps[8][16][32];  // P tiles per wave,   8 KB

  FragBF aq; // Q A-fragment (16 i-rows x K=32), loaded once
  {
    const uint16_t* qr = Qt + ((size_t)b * 4096 + i0 + wave * 16 + l16) * 32;
    aq.q[0] = *(const uint4*)(qr + half * 8);
    aq.q[1] = *(const uint4*)(qr + 16 + half * 8);
  }

  v8f acc[16]; // 16 rows x 256 channels, fp32
  #pragma unroll
  for (int ct = 0; ct < 16; ++ct) acc[ct] = (v8f){};
  float mrun[8], lrun[8];
  #pragma unroll
  for (int r = 0; r < 8; ++r) { mrun[r] = -3.0e38f; lrun[r] = 0.f; }

  const uint16_t* Vb = Vt + ((size_t)b * 256) * 4096;

#if HAVE_TDM
  if (wave == 0)
    tdm_load_v_chunk(Vb, (uint32_t)(uintptr_t)&Vs[0][0][0]);
#endif

  #pragma unroll 1
  for (int jc = 0; jc < NCHUNK; ++jc) {
    const int j0 = jc * 32;

    // K B-fragments straight from global (rows are fragment rows)
    const uint16_t* kb = Kt + ((size_t)b * 4096 + j0) * 32;
    FragBF bk0, bk1;
    bk0.q[0] = *(const uint4*)(kb + (size_t)l16 * 32 + half * 8);
    bk0.q[1] = *(const uint4*)(kb + (size_t)l16 * 32 + 16 + half * 8);
    bk1.q[0] = *(const uint4*)(kb + (size_t)(16 + l16) * 32 + half * 8);
    bk1.q[1] = *(const uint4*)(kb + (size_t)(16 + l16) * 32 + 16 + half * 8);

    v8f zero = {};
    v8f s0 = __builtin_amdgcn_wmma_f32_16x16x32_bf16(false, aq.v, false, bk0.v, (short)0, zero, false, false);
    v8f s1 = __builtin_amdgcn_wmma_f32_16x16x32_bf16(false, aq.v, false, bk1.v, (short)0, zero, false, false);

    float alpha[8];
    #pragma unroll
    for (int r = 0; r < 8; ++r) {
      // row m = half*8 + r lives across the 16 lanes of this half
      float mx = fmaxf(s0[r], s1[r]);
      #pragma unroll
      for (int mk = 1; mk <= 8; mk <<= 1)
        mx = fmaxf(mx, __shfl_xor(mx, mk, 32));
      float mnew = fmaxf(mrun[r], mx);
      alpha[r] = __expf(mrun[r] - mnew);
      float p0 = __expf(s0[r] - mnew);
      float p1 = __expf(s1[r] - mnew);
      s0[r] = p0; s1[r] = p1;
      float rs = p0 + p1;
      #pragma unroll
      for (int mk = 1; mk <= 8; mk <<= 1)
        rs += __shfl_xor(rs, mk, 32);
      lrun[r] = lrun[r] * alpha[r] + rs;
      mrun[r] = mnew;
    }
    // P (C-layout fp32) -> LDS bf16 for A-layout reload (wave-private region)
    #pragma unroll
    for (int r = 0; r < 8; ++r) {
      int m = half * 8 + r;
      Ps[wave][m][l16]      = f2bf(s0[r]);
      Ps[wave][m][16 + l16] = f2bf(s1[r]);
    }

    __syncthreads(); // [B] all waves done with previous V buffer reads

#if HAVE_TDM
    if (wave == 0) {
      if (jc + 1 < NCHUNK) {
        tdm_load_v_chunk(Vb + (jc + 1) * 32,
                         (uint32_t)(uintptr_t)&Vs[(jc + 1) & 1][0][0]);
        __builtin_amdgcn_s_wait_tensorcnt(1); // chunk jc complete, jc+1 in flight
      } else {
        __builtin_amdgcn_s_wait_tensorcnt(0); // final drain
      }
    }
#else
    { // synchronous fallback: stage chunk jc
      const uint4* src = (const uint4*)(Vb + (size_t)t * 4096 + j0);
      uint4* dst = (uint4*)&Vs[jc & 1][t][0];
      dst[0] = src[0]; dst[1] = src[1]; dst[2] = src[2]; dst[3] = src[3];
    }
#endif

    __syncthreads(); // [A] V chunk jc visible to all waves

    FragBF ap;
    ap.q[0] = *(const uint4*)&Ps[wave][l16][half * 8];
    ap.q[1] = *(const uint4*)&Ps[wave][l16][16 + half * 8];

    // rescale running output by alpha (same (r,half)->row mapping as C-layout)
    #pragma unroll
    for (int ct = 0; ct < 16; ++ct)
      #pragma unroll
      for (int r = 0; r < 8; ++r)
        acc[ct][r] *= alpha[r];

    #pragma unroll
    for (int ct = 0; ct < 16; ++ct) {
      FragBF bv_;
      bv_.q[0] = *(const uint4*)&Vs[jc & 1][ct * 16 + l16][half * 8];
      bv_.q[1] = *(const uint4*)&Vs[jc & 1][ct * 16 + l16][16 + half * 8];
      acc[ct] = __builtin_amdgcn_wmma_f32_16x16x32_bf16(false, ap.v, false, bv_.v, (short)0, acc[ct], false, false);
    }
  }

  // epilogue: out = gamma * O / l + x   (8 consecutive i per lane -> float4 x2)
  const float g = gamma_p[0];
  const int ibase = i0 + wave * 16 + half * 8;
  float inv_l[8];
  #pragma unroll
  for (int r = 0; r < 8; ++r) inv_l[r] = 1.f / lrun[r];
  #pragma unroll
  for (int ct = 0; ct < 16; ++ct) {
    int c = ct * 16 + l16;
    size_t base = ((size_t)b * 256 + c) * 4096 + ibase;
    float4 x0 = *(const float4*)&x[base];
    float4 x1 = *(const float4*)&x[base + 4];
    float4 o0, o1;
    o0.x = fmaf(g, acc[ct][0] * inv_l[0], x0.x);
    o0.y = fmaf(g, acc[ct][1] * inv_l[1], x0.y);
    o0.z = fmaf(g, acc[ct][2] * inv_l[2], x0.z);
    o0.w = fmaf(g, acc[ct][3] * inv_l[3], x0.w);
    o1.x = fmaf(g, acc[ct][4] * inv_l[4], x1.x);
    o1.y = fmaf(g, acc[ct][5] * inv_l[5], x1.y);
    o1.z = fmaf(g, acc[ct][6] * inv_l[6], x1.z);
    o1.w = fmaf(g, acc[ct][7] * inv_l[7], x1.w);
    *(float4*)&out[base]     = o0;
    *(float4*)&out[base + 4] = o1;
  }
}

// ---------------------------------------------------------------------------
extern "C" void kernel_launch(void* const* d_in, const int* in_sizes, int n_in,
                              void* d_out, int out_size, void* d_ws, size_t ws_size,
                              hipStream_t stream) {
  (void)in_sizes; (void)n_in; (void)out_size; (void)ws_size;
  const float* x     = (const float*)d_in[0];
  const float* wq    = (const float*)d_in[1];
  const float* bq    = (const float*)d_in[2];
  const float* wk    = (const float*)d_in[3];
  const float* bk    = (const float*)d_in[4];
  const float* wv    = (const float*)d_in[5];
  const float* bv    = (const float*)d_in[6];
  const float* gamma = (const float*)d_in[7];
  float* out = (float*)d_out;

  // workspace: Qt (2 MB) | Kt (2 MB) | Vt (16 MB), all bf16
  uint16_t* Qt = (uint16_t*)d_ws;
  uint16_t* Kt = Qt + (size_t)8 * 4096 * 32;
  uint16_t* Vt = Kt + (size_t)8 * 4096 * 32;

  qk_conv_kernel<<<dim3(64, 8), 256, 0, stream>>>(x, wq, bq, wk, bk, Qt, Kt);
  v_conv_wmma_kernel<<<dim3(128, 2, 8), 256, 0, stream>>>(x, wv, bv, Vt);
  attn_kernel<<<dim3(32, 8), 256, 0, stream>>>(x, gamma, Qt, Kt, Vt, out);
}
